// GenScore_11063835754636
// MI455X (gfx1250) — matrix operand order repaired and checked by reading, not compiled, not run on hardware
//
#include <hip/hip_runtime.h>

#define NSRC 20000
#define NTGT 20000
#define NEDGE 320000
#define DD 128
#define ES 264   // LDS row stride for edge_in (>=257, multiple of 8 -> 16B aligned chunks)

typedef __bf16 bf16_t;
typedef __attribute__((ext_vector_type(16))) __bf16 v16bf;
typedef __attribute__((ext_vector_type(8)))  __bf16 v8bf;
typedef __attribute__((ext_vector_type(8)))  float  v8f;

// ---- WMMA helpers (CDNA5 16x16x32 bf16, f32 accum) ----
__device__ __forceinline__ v16bf load_b16(const bf16_t* p) {
  union { v16bf v; v8bf h[2]; } u;
  u.h[0] = *(const v8bf*)(p);
  u.h[1] = *(const v8bf*)(p + 8);
  return u.v;
}
// A fragment: lane holds row m; 16 contiguous bf16 at kc+hi*16
__device__ __forceinline__ v16bf load_a16(const bf16_t* row, int hi, int kc) {
  union { v16bf v; v8bf h[2]; } u;
  u.h[0] = *(const v8bf*)(row + kc + hi * 16);
  u.h[1] = *(const v8bf*)(row + kc + hi * 16 + 8);
  return u.v;
}
__device__ __forceinline__ v8f wmma_bf16(v16bf a, v16bf b, v8f c) {
  return __builtin_amdgcn_wmma_f32_16x16x32_bf16(false, a, false, b, (short)0, c,
                                                 false, false);
}
__device__ __forceinline__ float silu_f(float x) { return x / (1.0f + __expf(-x)); }

// ======================= edge kernel =======================
// 16 edges per block, 256 threads (8 wave32). Entire edge pipeline fused:
// gather -> MLP1 (K=257) -> MLP2 -> gate -> coord MLP -> scatter atomics.
__global__ __launch_bounds__(256) void egnn_edge_kernel(
    const float* __restrict__ srcf, const float* __restrict__ tgtf,
    const float* __restrict__ srcc, const float* __restrict__ tgtc,
    const int* __restrict__ e_src, const int* __restrict__ e_tgt,
    const bf16_t* __restrict__ w1T_s2t, const float* __restrict__ b1_s2t,
    const bf16_t* __restrict__ w2T_s2t, const float* __restrict__ b2_s2t,
    const bf16_t* __restrict__ w1T_t2s, const float* __restrict__ b1_t2s,
    const bf16_t* __restrict__ w2T_t2s, const float* __restrict__ b2_t2s,
    const float* __restrict__ wg_s2t, const float* __restrict__ bg_s2t,
    const float* __restrict__ wg_t2s, const float* __restrict__ bg_t2s,
    const bf16_t* __restrict__ wc1T, const float* __restrict__ bc1,
    const float* __restrict__ wc2,
    float* __restrict__ agg_s2t, float* __restrict__ agg_t2s,
    float* __restrict__ trans_sum, float* __restrict__ cnt)
{
  __shared__ __align__(16) bf16_t s_in[16 * ES];
  __shared__ __align__(16) bf16_t s_h[2][16 * DD];
  __shared__ __align__(16) bf16_t s_e[2][16 * DD];
  __shared__ __align__(16) bf16_t s_ch[16 * DD];
  __shared__ float s_cd[16][3];
  __shared__ float s_rad[16];
  __shared__ float s_g[2][16];
  __shared__ float s_cw[16];
  __shared__ int   s_es[16], s_et[16];

  const int tid  = threadIdx.x;
  const int lane = tid & 31;
  const int wave = tid >> 5;
  const int mcol = lane & 15;  // A row / B,C column within tile
  const int hi   = lane >> 4;
  const int e0   = blockIdx.x * 16;

  // indices, coord_diff, radial
  if (tid < 16) {
    int es = e_src[e0 + tid], et = e_tgt[e0 + tid];
    s_es[tid] = es; s_et[tid] = et;
    float dx = tgtc[et * 3 + 0] - srcc[es * 3 + 0];
    float dy = tgtc[et * 3 + 1] - srcc[es * 3 + 1];
    float dz = tgtc[et * 3 + 2] - srcc[es * 3 + 2];
    s_cd[tid][0] = dx; s_cd[tid][1] = dy; s_cd[tid][2] = dz;
    s_rad[tid] = dx * dx + dy * dy + dz * dz;
  }
  __syncthreads();

  // gather edge_in (bf16) into LDS: [16 x 257]
  for (int idx = tid; idx < 16 * 257; idx += 256) {
    int e = idx / 257, c = idx - e * 257;
    float v;
    if (c < DD)           v = srcf[(size_t)s_es[e] * DD + c];
    else if (c < 2 * DD)  v = tgtf[(size_t)s_et[e] * DD + (c - DD)];
    else                  v = s_rad[e];
    s_in[e * ES + c] = (bf16_t)v;
  }
  __syncthreads();

  // ---- edge MLP layer 1: 16 output tiles (2 MLPs x 8), 2 tiles per wave ----
  for (int r = 0; r < 2; ++r) {
    int t = wave * 2 + r;
    int mlp = t >> 3, nt = t & 7;
    const bf16_t* wT = mlp ? w1T_t2s : w1T_s2t;
    const float*  b1 = mlp ? b1_t2s : b1_s2t;
    v8f acc = {};
    for (int kc = 0; kc < 256; kc += 32) {
      v16bf a = load_a16(s_in + mcol * ES, hi, kc);
      v16bf b = load_b16(wT + (size_t)(nt * 16 + mcol) * ES + kc + hi * 16);
      acc = wmma_bf16(a, b, acc);
    }
    int col = nt * 16 + mcol;
    float rw = (float)wT[(size_t)col * ES + 256];  // radial column (K=256)
    float bias = b1[col];
#pragma unroll
    for (int i = 0; i < 8; ++i) {
      int mm = hi * 8 + i;
      float x = acc[i] + s_rad[mm] * rw + bias;
      s_h[mlp][mm * DD + col] = (bf16_t)silu_f(x);
    }
  }
  __syncthreads();

  // ---- edge MLP layer 2 ----
  for (int r = 0; r < 2; ++r) {
    int t = wave * 2 + r;
    int mlp = t >> 3, nt = t & 7;
    const bf16_t* wT = mlp ? w2T_t2s : w2T_s2t;
    const float*  b2 = mlp ? b2_t2s : b2_s2t;
    v8f acc = {};
    for (int kc = 0; kc < DD; kc += 32) {
      v16bf a = load_a16(s_h[mlp] + mcol * DD, hi, kc);
      v16bf b = load_b16(wT + (size_t)(nt * 16 + mcol) * DD + kc + hi * 16);
      acc = wmma_bf16(a, b, acc);
    }
    int col = nt * 16 + mcol;
    float bias = b2[col];
#pragma unroll
    for (int i = 0; i < 8; ++i) {
      int mm = hi * 8 + i;
      s_e[mlp][mm * DD + col] = (bf16_t)silu_f(acc[i] + bias);
    }
  }
  __syncthreads();

  // ---- gates: g = sigmoid(e . wg + bg), then e *= g ----
  if (tid < 32) {
    int e = tid & 15, mlp = tid >> 4;
    const float* wg = mlp ? wg_t2s : wg_s2t;
    float bg = mlp ? bg_t2s[0] : bg_s2t[0];
    float dot = 0.f;
    for (int k = 0; k < DD; ++k) dot += (float)s_e[mlp][e * DD + k] * wg[k];
    s_g[mlp][e] = 1.f / (1.f + __expf(-(dot + bg)));
  }
  __syncthreads();
  for (int idx = tid; idx < 2 * 16 * DD; idx += 256) {
    int mlp = idx >> 11, rem = idx & 2047;
    int e = rem >> 7, k = rem & 127;
    s_e[mlp][e * DD + k] = (bf16_t)((float)s_e[mlp][e * DD + k] * s_g[mlp][e]);
  }
  __syncthreads();

  // ---- coord MLP: ch = silu(e_s2t @ wc1 + bc1); 8 tiles, 1 per wave ----
  {
    int nt = wave;
    v8f acc = {};
    for (int kc = 0; kc < DD; kc += 32) {
      v16bf a = load_a16(s_e[0] + mcol * DD, hi, kc);
      v16bf b = load_b16(wc1T + (size_t)(nt * 16 + mcol) * DD + kc + hi * 16);
      acc = wmma_bf16(a, b, acc);
    }
    int col = nt * 16 + mcol;
    float bias = bc1[col];
#pragma unroll
    for (int i = 0; i < 8; ++i) {
      int mm = hi * 8 + i;
      s_ch[mm * DD + col] = (bf16_t)silu_f(acc[i] + bias);
    }
  }
  __syncthreads();
  if (tid < 16) {
    float dot = 0.f;
    for (int k = 0; k < DD; ++k) dot += (float)s_ch[tid * DD + k] * wc2[k];
    s_cw[tid] = dot;
  }
  __syncthreads();

  // ---- scatter: feature aggregation + weighted coord translation + counts ----
  for (int idx = tid; idx < 16 * DD; idx += 256) {
    int e = idx >> 7, k = idx & 127;
    unsafeAtomicAdd(&agg_s2t[(size_t)s_et[e] * DD + k], (float)s_e[0][e * DD + k]);
    unsafeAtomicAdd(&agg_t2s[(size_t)s_es[e] * DD + k], (float)s_e[1][e * DD + k]);
  }
  if (tid < 48) {
    int e = tid / 3, d = tid - e * 3;
    unsafeAtomicAdd(&trans_sum[(size_t)s_et[e] * 3 + d], s_cd[e][d] * s_cw[e]);
  }
  if (tid < 16) unsafeAtomicAdd(&cnt[s_et[tid]], 1.0f);
}

// ======================= node kernel =======================
// out = feat + (silu([feat|agg] @ wn1 + b1) @ wn2 + b2); 16 nodes/block.
__global__ __launch_bounds__(256) void egnn_node_kernel(
    const float* __restrict__ feat, const float* __restrict__ agg,
    const bf16_t* __restrict__ wn1T, const float* __restrict__ bn1,
    const bf16_t* __restrict__ wn2T, const float* __restrict__ bn2,
    float* __restrict__ out)
{
  __shared__ __align__(16) bf16_t s_in[16 * 256];
  __shared__ __align__(16) bf16_t s_h[16 * DD];
  const int tid  = threadIdx.x;
  const int lane = tid & 31;
  const int wave = tid >> 5;
  const int mcol = lane & 15;
  const int hi   = lane >> 4;
  const int n0   = blockIdx.x * 16;

  for (int idx = tid; idx < 16 * 256; idx += 256) {
    int r = idx >> 8, c = idx & 255;
    float v = (c < DD) ? feat[(size_t)(n0 + r) * DD + c]
                       : agg[(size_t)(n0 + r) * DD + (c - DD)];
    s_in[r * 256 + c] = (bf16_t)v;
  }
  __syncthreads();

  {  // layer 1: K = 256
    int nt = wave;
    v8f acc = {};
    for (int kc = 0; kc < 256; kc += 32) {
      v16bf a = load_a16(s_in + mcol * 256, hi, kc);
      v16bf b = load_b16(wn1T + (size_t)(nt * 16 + mcol) * 256 + kc + hi * 16);
      acc = wmma_bf16(a, b, acc);
    }
    int col = nt * 16 + mcol;
    float bias = bn1[col];
#pragma unroll
    for (int i = 0; i < 8; ++i) {
      int mm = hi * 8 + i;
      s_h[mm * DD + col] = (bf16_t)silu_f(acc[i] + bias);
    }
  }
  __syncthreads();

  {  // layer 2: K = 128, + bias + residual
    int nt = wave;
    v8f acc = {};
    for (int kc = 0; kc < DD; kc += 32) {
      v16bf a = load_a16(s_h + mcol * DD, hi, kc);
      v16bf b = load_b16(wn2T + (size_t)(nt * 16 + mcol) * DD + kc + hi * 16);
      acc = wmma_bf16(a, b, acc);
    }
    int col = nt * 16 + mcol;
    float bias = bn2[col];
#pragma unroll
    for (int i = 0; i < 8; ++i) {
      int mm = hi * 8 + i;
      size_t o = (size_t)(n0 + mm) * DD + col;
      out[o] = feat[o] + acc[i] + bias;
    }
  }
}

// ---- small utility kernels ----
__global__ void convT_kernel(const float* __restrict__ W, bf16_t* __restrict__ WT,
                             int K, int N, int stride) {
  int idx = blockIdx.x * blockDim.x + threadIdx.x;
  if (idx < K * N) {
    int k = idx / N, n = idx - k * N;
    WT[(size_t)n * stride + k] = (bf16_t)W[idx];
  }
}
__global__ void copyf_kernel(const float* __restrict__ in, float* __restrict__ out, int n) {
  int i = blockIdx.x * blockDim.x + threadIdx.x;
  if (i < n) out[i] = in[i];
}
__global__ void coord_final_kernel(const float* __restrict__ tgtc,
                                   const float* __restrict__ trans,
                                   const float* __restrict__ cnt,
                                   float* __restrict__ out, int n3) {
  int i = blockIdx.x * blockDim.x + threadIdx.x;
  if (i < n3) {
    int node = i / 3;
    float c = fmaxf(cnt[node], 1.0f);
    float a = trans[i] / c;
    a = fminf(fmaxf(a, -10.0f), 10.0f);
    out[i] = tgtc[i] + a;
  }
}

extern "C" void kernel_launch(void* const* d_in, const int* in_sizes, int n_in,
                              void* d_out, int out_size, void* d_ws, size_t ws_size,
                              hipStream_t stream) {
  const float* srcf = (const float*)d_in[0];
  const float* tgtf = (const float*)d_in[1];
  const float* srcc = (const float*)d_in[2];
  const float* tgtc = (const float*)d_in[3];
  const int*   e_src = (const int*)d_in[4];
  const int*   e_tgt = (const int*)d_in[5];
  const float* w1_s2t = (const float*)d_in[6];  const float* b1_s2t = (const float*)d_in[7];
  const float* w2_s2t = (const float*)d_in[8];  const float* b2_s2t = (const float*)d_in[9];
  const float* w1_t2s = (const float*)d_in[10]; const float* b1_t2s = (const float*)d_in[11];
  const float* w2_t2s = (const float*)d_in[12]; const float* b2_t2s = (const float*)d_in[13];
  const float* wg_s2t = (const float*)d_in[14]; const float* bg_s2t = (const float*)d_in[15];
  const float* wg_t2s = (const float*)d_in[16]; const float* bg_t2s = (const float*)d_in[17];
  const float* wc1 = (const float*)d_in[18];    const float* bc1 = (const float*)d_in[19];
  const float* wc2 = (const float*)d_in[20];
  const float* wn1_t = (const float*)d_in[21];  const float* bn1_t = (const float*)d_in[22];
  const float* wn2_t = (const float*)d_in[23];  const float* bn2_t = (const float*)d_in[24];
  const float* wn1_s = (const float*)d_in[25];  const float* bn1_s = (const float*)d_in[26];
  const float* wn2_s = (const float*)d_in[27];  const float* bn2_s = (const float*)d_in[28];

  // workspace carve-out
  char* ws = (char*)d_ws;
  float* agg_s2t  = (float*)ws; ws += (size_t)NTGT * DD * 4;
  float* agg_t2s  = (float*)ws; ws += (size_t)NSRC * DD * 4;
  float* trans_sum = (float*)ws; ws += (size_t)NTGT * 3 * 4;
  float* cnt      = (float*)ws; ws += (size_t)NTGT * 4;
  size_t zero_bytes = (size_t)(ws - (char*)d_ws);
  bf16_t* w1T_s2t = (bf16_t*)ws; ws += (size_t)DD * ES * 2;
  bf16_t* w1T_t2s = (bf16_t*)ws; ws += (size_t)DD * ES * 2;
  bf16_t* w2T_s2t = (bf16_t*)ws; ws += (size_t)DD * DD * 2;
  bf16_t* w2T_t2s = (bf16_t*)ws; ws += (size_t)DD * DD * 2;
  bf16_t* wc1T    = (bf16_t*)ws; ws += (size_t)DD * DD * 2;
  bf16_t* wn1T_t  = (bf16_t*)ws; ws += (size_t)DD * 256 * 2;
  bf16_t* wn2T_t  = (bf16_t*)ws; ws += (size_t)DD * DD * 2;
  bf16_t* wn1T_s  = (bf16_t*)ws; ws += (size_t)DD * 256 * 2;
  bf16_t* wn2T_s  = (bf16_t*)ws; ws += (size_t)DD * DD * 2;

  hipMemsetAsync(d_ws, 0, zero_bytes, stream);

  // weight transpose + fp32 -> bf16
  auto conv = [&](const float* W, bf16_t* WT, int K, int N, int stride) {
    int total = K * N;
    convT_kernel<<<(total + 255) / 256, 256, 0, stream>>>(W, WT, K, N, stride);
  };
  conv(w1_s2t, w1T_s2t, 257, DD, ES);
  conv(w1_t2s, w1T_t2s, 257, DD, ES);
  conv(w2_s2t, w2T_s2t, DD, DD, DD);
  conv(w2_t2s, w2T_t2s, DD, DD, DD);
  conv(wc1,    wc1T,    DD, DD, DD);
  conv(wn1_t,  wn1T_t,  256, DD, 256);
  conv(wn2_t,  wn2T_t,  DD, DD, DD);
  conv(wn1_s,  wn1T_s,  256, DD, 256);
  conv(wn2_s,  wn2T_s,  DD, DD, DD);

  // fused edge pipeline
  egnn_edge_kernel<<<NEDGE / 16, 256, 0, stream>>>(
      srcf, tgtf, srcc, tgtc, e_src, e_tgt,
      w1T_s2t, b1_s2t, w2T_s2t, b2_s2t,
      w1T_t2s, b1_t2s, w2T_t2s, b2_t2s,
      wg_s2t, bg_s2t, wg_t2s, bg_t2s,
      wc1T, bc1, wc2,
      agg_s2t, agg_t2s, trans_sum, cnt);

  // outputs: [src_out | tgt_out | src_coord | tgt_coord]
  float* out = (float*)d_out;
  float* src_out = out;
  float* tgt_out = out + (size_t)NSRC * DD;
  float* src_coord_out = tgt_out + (size_t)NTGT * DD;
  float* tgt_coord_out = src_coord_out + (size_t)NSRC * 3;

  egnn_node_kernel<<<NTGT / 16, 256, 0, stream>>>(tgtf, agg_s2t, wn1T_t, bn1_t,
                                                  wn2T_t, bn2_t, tgt_out);
  egnn_node_kernel<<<NSRC / 16, 256, 0, stream>>>(srcf, agg_t2s, wn1T_s, bn1_s,
                                                  wn2T_s, bn2_s, src_out);
  copyf_kernel<<<(NSRC * 3 + 255) / 256, 256, 0, stream>>>(srcc, src_coord_out, NSRC * 3);
  coord_final_kernel<<<(NTGT * 3 + 255) / 256, 256, 0, stream>>>(tgtc, trans_sum, cnt,
                                                                 tgt_coord_out, NTGT * 3);
}
